// AttentionModel_91018946937107
// MI455X (gfx1250) — matrix operand order repaired and checked by reading, not compiled, not run on hardware
//
#include <hip/hip_runtime.h>
#include <hip/hip_bf16.h>

// ---------------------------------------------------------------------------
// Problem constants (from reference): b=8, n=64, r=256, d=512
//   x  : [512][256][512] fp32 (bn-flattened)
//   W1 : [512][512] fp32, b1: [512], W2: [512], b2: scalar (no effect on outputs)
//   out: fbar [512][512] fp32  followed by  alphas [512][256] fp32
// ---------------------------------------------------------------------------

typedef __attribute__((ext_vector_type(16))) __bf16 v16bf;
typedef __attribute__((ext_vector_type(8)))  float  v8f;

#define NREG   256          // regions per (b,n)
#define DDIM   512          // feature dim
#define NBN    512          // b*n workgroups
#define FBAR_ELEMS (NBN * DDIM)   // 262144 floats, alphas follow

#define LDS_XN_BYTES (NREG * DDIM * 2)              // 262144: bf16 Xn
#define LDS_B1_OFF   (LDS_XN_BYTES)                 // 512 floats
#define LDS_W2_OFF   (LDS_B1_OFF + DDIM * 4)        // 512 floats
#define LDS_SC_OFF   (LDS_W2_OFF + DDIM * 4)        // 256 floats (scores)
#define LDS_AL_OFF   (LDS_SC_OFF + NREG * 4)        // 256 floats (alphas)
#define LDS_TOTAL    (LDS_AL_OFF + NREG * 4)        // 268288 B  (< 320 KB WGP LDS)

__device__ __forceinline__ unsigned int f2bf_rne(float f) {
  unsigned int u = __float_as_uint(f);
  return (u + 0x7FFFu + ((u >> 16) & 1u)) >> 16;      // round-to-nearest-even
}
__device__ __forceinline__ unsigned int pack2bf(float lo, float hi) {
  return f2bf_rne(lo) | (f2bf_rne(hi) << 16);
}
__device__ __forceinline__ float bflo(unsigned int p) { return __uint_as_float(p << 16); }
__device__ __forceinline__ float bfhi(unsigned int p) { return __uint_as_float(p & 0xFFFF0000u); }

union Frag16 { uint4 u[2]; v16bf v; };
__device__ __forceinline__ v16bf make_frag(uint4 a, uint4 b) {
  Frag16 f; f.u[0] = a; f.u[1] = b; return f.v;
}

// --------------------------- W1 fp32 -> bf16 -------------------------------
__global__ void __launch_bounds__(256, 1)
w1_to_bf16_kernel(const float* __restrict__ W1, unsigned int* __restrict__ w1q) {
  int i = blockIdx.x * 256 + threadIdx.x;             // packs 2 elements
  w1q[i] = pack2bf(W1[2 * i], W1[2 * i + 1]);
}

// ------------------------------ main kernel --------------------------------
__global__ void __launch_bounds__(256, 1)
attn_fused_kernel(const float* __restrict__ x,
                  const unsigned int* __restrict__ w1bf,   // bf16 pairs [512][256]
                  const float* __restrict__ b1,
                  const float* __restrict__ W2,
                  float* __restrict__ out) {
  extern __shared__ unsigned char smem[];
  unsigned int* xnd   = (unsigned int*)smem;               // [256][256] bf16 pairs
  float*        b1s   = (float*)(smem + LDS_B1_OFF);
  float*        w2s   = (float*)(smem + LDS_W2_OFF);
  float*        score = (float*)(smem + LDS_SC_OFF);
  float*        alpha = (float*)(smem + LDS_AL_OFF);

  const int tid  = threadIdx.x;
  const int lane = tid & 31;
  const int wave = tid >> 5;          // 8 waves (wave32)
  const int lo16 = lane & 15;
  const int hi   = lane >> 4;
  const int bn   = blockIdx.x;
  const size_t xbase = (size_t)bn * (NREG * DDIM);

  // stage b1/W2, zero score accumulators
  b1s[tid] = b1[tid];  b1s[tid + 256] = b1[tid + 256];
  w2s[tid] = W2[tid];  w2s[tid + 256] = W2[tid + 256];
  score[tid] = 0.0f;

  // ---- Phase 1: normalize rows, write bf16 Xn to LDS (wave w: rows w*32..) ----
  for (int rr = 0; rr < 32; ++rr) {
    const int row = wave * 32 + rr;
    const float4* xr = (const float4*)(x + xbase + (size_t)row * DDIM);
    float4 a0 = xr[lane], a1 = xr[lane + 32], a2 = xr[lane + 64], a3 = xr[lane + 96];
    float s = a0.x*a0.x + a0.y*a0.y + a0.z*a0.z + a0.w*a0.w
            + a1.x*a1.x + a1.y*a1.y + a1.z*a1.z + a1.w*a1.w
            + a2.x*a2.x + a2.y*a2.y + a2.z*a2.z + a2.w*a2.w
            + a3.x*a3.x + a3.y*a3.y + a3.z*a3.z + a3.w*a3.w;
    s += __shfl_xor(s, 1, 32);  s += __shfl_xor(s, 2, 32);
    s += __shfl_xor(s, 4, 32);  s += __shfl_xor(s, 8, 32);
    s += __shfl_xor(s, 16, 32);
    const float scale = 1.0f / fmaxf(sqrtf(s), 1e-12f);
    uint2* dst = (uint2*)smem;
    const int base = row * 128;
    dst[base + lane     ] = make_uint2(pack2bf(a0.x*scale, a0.y*scale), pack2bf(a0.z*scale, a0.w*scale));
    dst[base + lane + 32] = make_uint2(pack2bf(a1.x*scale, a1.y*scale), pack2bf(a1.z*scale, a1.w*scale));
    dst[base + lane + 64] = make_uint2(pack2bf(a2.x*scale, a2.y*scale), pack2bf(a2.z*scale, a2.w*scale));
    dst[base + lane + 96] = make_uint2(pack2bf(a3.x*scale, a3.y*scale), pack2bf(a3.z*scale, a3.w*scale));
  }
  __syncthreads();

  // ---- Phase 2: H = tanh(Xn*W1^T + b1), fused score = H . W2 ----
  // Wave owns 64 output cols; two 32-col passes with B register-resident so
  // W1 is read exactly once per workgroup (L2-resident bf16 copy).
  const uint4* xnq = (const uint4*)smem;   // A fragments (LDS)
  const uint4* w1q = (const uint4*)w1bf;   // B fragments (global/L2)

  for (int pass = 0; pass < 2; ++pass) {
    const int colBase = wave * 64 + pass * 32;
    v16bf Bfrag[2][16];                    // 2 ntiles x 16 kfrags = 256 VGPRs
    float b1v[2], w2v[2];
#pragma unroll
    for (int t = 0; t < 2; ++t) {
      const int n = colBase + t * 16 + lo16;           // B lane column
      b1v[t] = b1s[n];
      w2v[t] = w2s[n];
      // B[k][n] = W1[n][k]; lane holds k = kk*32 + hi*16 + 0..15 (2 per VGPR)
      const uint4* rowp = w1q + n * 64 + hi * 2;
#pragma unroll
      for (int kk = 0; kk < 16; ++kk)
        Bfrag[t][kk] = make_frag(rowp[kk * 4], rowp[kk * 4 + 1]);
    }

    for (int mTile = 0; mTile < 16; ++mTile) {
      v8f acc0 = {}; v8f acc1 = {};
      // A lane row m = mTile*16 + lo16; k runs: kk*32 + hi*16 + 0..15
      const uint4* arow = xnq + (mTile * 16 + lo16) * 64 + hi * 2;
#pragma unroll
      for (int kk = 0; kk < 16; ++kk) {
        v16bf a = make_frag(arow[kk * 4], arow[kk * 4 + 1]);
        acc0 = __builtin_amdgcn_wmma_f32_16x16x32_bf16(false, a, false, Bfrag[0][kk],
                                                       (short)0, acc0, false, false);
        acc1 = __builtin_amdgcn_wmma_f32_16x16x32_bf16(false, a, false, Bfrag[1][kk],
                                                       (short)0, acc1, false, false);
      }
      // Epilogue: tanh + W2 dot; D layout: VGPR v -> M = v + 8*hi, N = lo16
      const int rowBase = mTile * 16 + hi * 8;
#pragma unroll
      for (int v = 0; v < 8; ++v) {
        float c = tanhf(acc0[v] + b1v[0]) * w2v[0]
                + tanhf(acc1[v] + b1v[1]) * w2v[1];
        c += __shfl_xor(c, 1, 32);
        c += __shfl_xor(c, 2, 32);
        c += __shfl_xor(c, 4, 32);
        c += __shfl_xor(c, 8, 32);
        if (lo16 == 0) atomicAdd(&score[rowBase + v], c);   // ds_add_f32
      }
    }
  }
  __syncthreads();

  // ---- Phase 3: softmax over 256 regions (wave 0). b2 is shift-invariant. ----
  if (wave == 0) {
    float sc[8];
    float m = -3.402823466e38f;
#pragma unroll
    for (int i = 0; i < 8; ++i) { sc[i] = score[lane * 8 + i]; m = fmaxf(m, sc[i]); }
    m = fmaxf(m, __shfl_xor(m, 1, 32));  m = fmaxf(m, __shfl_xor(m, 2, 32));
    m = fmaxf(m, __shfl_xor(m, 4, 32));  m = fmaxf(m, __shfl_xor(m, 8, 32));
    m = fmaxf(m, __shfl_xor(m, 16, 32));
    float sum = 0.0f;
#pragma unroll
    for (int i = 0; i < 8; ++i) { sc[i] = __expf(sc[i] - m); sum += sc[i]; }
    sum += __shfl_xor(sum, 1, 32);  sum += __shfl_xor(sum, 2, 32);
    sum += __shfl_xor(sum, 4, 32);  sum += __shfl_xor(sum, 8, 32);
    sum += __shfl_xor(sum, 16, 32);
    const float inv = 1.0f / sum;
    float* outAl = out + FBAR_ELEMS + (size_t)bn * NREG;
#pragma unroll
    for (int i = 0; i < 8; ++i) {
      const float a = sc[i] * inv;
      alpha[lane * 8 + i] = a;
      outAl[lane * 8 + i] = a;
    }
  }
  __syncthreads();

  // ---- Phase 4: fbar = sum_r alpha_r * Xn_r (thread owns cols 2t, 2t+1) ----
  float f0 = 0.0f, f1 = 0.0f;
  for (int r = 0; r < NREG; ++r) {
    const float a = alpha[r];
    const unsigned int p = xnd[r * 256 + tid];
    f0 += a * bflo(p);
    f1 += a * bfhi(p);
  }
  ((float2*)out)[(size_t)bn * 256 + tid] = make_float2(f0, f1);
}

// ------------------------------- launcher ----------------------------------
extern "C" void kernel_launch(void* const* d_in, const int* in_sizes, int n_in,
                              void* d_out, int out_size, void* d_ws, size_t ws_size,
                              hipStream_t stream) {
  (void)in_sizes; (void)n_in; (void)out_size; (void)ws_size;
  const float* x  = (const float*)d_in[0];
  const float* W1 = (const float*)d_in[1];
  const float* b1 = (const float*)d_in[2];
  const float* W2 = (const float*)d_in[3];
  // d_in[4] (b2) intentionally unused: softmax(scores + b2) == softmax(scores)
  // and raw scores are not part of the outputs.
  unsigned int* w1bf = (unsigned int*)d_ws;   // 512 KB bf16 copy of W1

  w1_to_bf16_kernel<<<512, 256, 0, stream>>>(W1, w1bf);

  static_assert(LDS_TOTAL <= 320 * 1024, "LDS budget");
  hipFuncSetAttribute((const void*)attn_fused_kernel,
                      hipFuncAttributeMaxDynamicSharedMemorySize, LDS_TOTAL);
  attn_fused_kernel<<<NBN, 256, LDS_TOTAL, stream>>>(x, w1bf, b1, W2, (float*)d_out);
}
